// RNNModel_10969346474402
// MI455X (gfx1250) — matrix operand order repaired and checked by reading, not compiled, not run on hardware
//
#include <hip/hip_runtime.h>
#include <hip/hip_bf16.h>

typedef __attribute__((ext_vector_type(16))) _Float16 v16h;
typedef __attribute__((ext_vector_type(2)))  _Float16 h2;
typedef __attribute__((ext_vector_type(8)))  float    v8f;

constexpr int kH   = 64;    // hidden size
constexpr int kT   = 256;   // sequence length
constexpr int kB   = 8192;  // batch
constexpr int kRows = 16;   // batch rows per wave (WMMA M)
constexpr int kHSH = 68;    // h-state LDS row stride in halves (even, bank-padded)

// ---- activations: prefer native v_tanh_f32 (CDNA5 TRANS op) ----
#if __has_builtin(__builtin_amdgcn_tanhf)
__device__ __forceinline__ float fast_tanh(float x) {
    return __builtin_amdgcn_tanhf(x);
}
__device__ __forceinline__ float fast_sigmoid(float x) {
    // sigmoid(x) = 0.5 + 0.5*tanh(0.5*x)
    return __builtin_fmaf(0.5f, __builtin_amdgcn_tanhf(0.5f * x), 0.5f);
}
#else
__device__ __forceinline__ float fast_sigmoid(float x) {
    float e = __builtin_amdgcn_exp2f(-1.4426950408889634f * x);
    return __builtin_amdgcn_rcpf(1.0f + e);
}
__device__ __forceinline__ float fast_tanh(float x) {
    float e = __builtin_amdgcn_exp2f(-2.8853900817779268f * x);
    return 2.0f * __builtin_amdgcn_rcpf(1.0f + e) - 1.0f;
}
#endif

// acc init with bias + x*W_ih (C layout: row m = v + 8*hi, col = 16*nt + ln),
// then two K=32 WMMAs accumulate h @ W_hh^T for this 16x16 gate tile.
#define GATE_MM(acc, nt) do {                                                   \
    _Pragma("unroll")                                                           \
    for (int v = 0; v < 8; ++v) acc[v] = bias_r[nt] + xm[v] * wih_r[nt];        \
    acc = __builtin_amdgcn_wmma_f32_16x16x32_f16(false, A0, false, Bf[nt][0],   \
                                                 (short)0, acc, false, false);  \
    acc = __builtin_amdgcn_wmma_f32_16x16x32_f16(false, A1, false, Bf[nt][1],   \
                                                 (short)0, acc, false, false);  \
} while (0)

__global__ __launch_bounds__(32, 1)
void lstm_wmma_kernel(const float* __restrict__ x,
                      const float* __restrict__ W_ih,
                      const float* __restrict__ W_hh,
                      const float* __restrict__ b_ih,
                      const float* __restrict__ b_hh,
                      const float* __restrict__ fc1_w,
                      const float* __restrict__ fc1_b,
                      const float* __restrict__ fc2_w,
                      const float* __restrict__ fc2_b,
                      float* __restrict__ out)
{
    // Single-wave workgroup: barriers lower to S_NOP, LDS is wave-private.
    __shared__ _Float16 hbuf[kRows * kHSH];   // h state (f16)
    __shared__ float    zbuf[kRows * 32];     // fc1 activations

    const int lane = threadIdx.x & 31;
    const int hi   = lane >> 4;      // lane half (0/1)
    const int ln   = lane & 15;
    const int rowbase = blockIdx.x * kRows;

    // ---- per-lane gate constants: bias and W_ih column (col = 16*nt + ln) ----
    float bias_r[16], wih_r[16];
#pragma unroll
    for (int nt = 0; nt < 16; ++nt) {
        int g = 16 * nt + ln;
        bias_r[nt] = b_ih[g] + b_hh[g];
        wih_r[nt]  = W_ih[g];
    }

    // ---- W_hh^T as resident f16 B-fragments ----
    // Bf[nt][kh]: B[k][n], n = 16*nt + ln, k = 32*kh + 8*hi + 2v + (v>=4 ? 8 : 0) + {0,1}
    v16h Bf[16][2];
#pragma unroll
    for (int nt = 0; nt < 16; ++nt) {
#pragma unroll
        for (int kh = 0; kh < 2; ++kh) {
            int g = 16 * nt + ln;
            v16h bv = {};
#pragma unroll
            for (int v = 0; v < 8; ++v) {
                int k = 32 * kh + 8 * hi + 2 * v + (v >= 4 ? 8 : 0);
                bv[2 * v]     = (_Float16)W_hh[g * kH + k];
                bv[2 * v + 1] = (_Float16)W_hh[g * kH + k + 1];
            }
            Bf[nt][kh] = bv;
        }
    }

    // h0 = 0
    for (int i = lane; i < kRows * kHSH; i += 32) hbuf[i] = (_Float16)0.0f;
    __syncthreads();

    // c0 = 0 (same (tile j, VGPR v) layout as the gate tiles)
    float c_s[4][8];
#pragma unroll
    for (int j = 0; j < 4; ++j)
#pragma unroll
        for (int v = 0; v < 8; ++v) c_s[j][v] = 0.0f;

    const float* xrow = x + rowbase * kT;
    const h2* hpair = reinterpret_cast<const h2*>(hbuf);  // indexed in half-pairs

#pragma unroll 1
    for (int t = 0; t < kT; ++t) {
        // x for the 8 batch rows this lane accumulates (row m = v + 8*hi)
        float xm[8];
#pragma unroll
        for (int v = 0; v < 8; ++v)
            xm[v] = xrow[(v + 8 * hi) * kT + t];

        // Build A fragments (h_{t-1}, already f16 in LDS) — one packed-pair
        // b32 load per A VGPR. A layout: lane ln = row, VGPR v holds K0,K0+1,
        //   K0 = 2v + 8*hi + (v>=4 ? 8 : 0)   (per 16x32 fragment)
        v16h A0 = {}, A1 = {};
#pragma unroll
        for (int v = 0; v < 8; ++v) {
            int K0 = 2 * v + 8 * hi + (v >= 4 ? 8 : 0);
            h2 p0 = hpair[(ln * kHSH + K0) >> 1];
            h2 p1 = hpair[(ln * kHSH + 32 + K0) >> 1];
            A0[2 * v] = p0[0]; A0[2 * v + 1] = p0[1];
            A1[2 * v] = p1[0]; A1[2 * v + 1] = p1[1];
        }

        float gg[4][8];  // tanh(g), then sigmoid(i)*tanh(g)

        // gate g (tiles 8..11)
#pragma unroll
        for (int j = 0; j < 4; ++j) {
            v8f acc;
            GATE_MM(acc, 8 + j);
#pragma unroll
            for (int v = 0; v < 8; ++v) gg[j][v] = fast_tanh(acc[v]);
        }
        // gate i (tiles 0..3)
#pragma unroll
        for (int j = 0; j < 4; ++j) {
            v8f acc;
            GATE_MM(acc, j);
#pragma unroll
            for (int v = 0; v < 8; ++v) gg[j][v] *= fast_sigmoid(acc[v]);
        }
        // gate f (tiles 4..7): c = sigmoid(f)*c + i*g
#pragma unroll
        for (int j = 0; j < 4; ++j) {
            v8f acc;
            GATE_MM(acc, 4 + j);
#pragma unroll
            for (int v = 0; v < 8; ++v)
                c_s[j][v] = fast_sigmoid(acc[v]) * c_s[j][v] + gg[j][v];
        }
        // gate o (tiles 12..15): h = sigmoid(o)*tanh(c) -> LDS (f16)
#pragma unroll
        for (int j = 0; j < 4; ++j) {
            v8f acc;
            GATE_MM(acc, 12 + j);
#pragma unroll
            for (int v = 0; v < 8; ++v) {
                float hval = fast_sigmoid(acc[v]) * fast_tanh(c_s[j][v]);
                hbuf[(v + 8 * hi) * kHSH + 16 * j + ln] = (_Float16)hval;
            }
        }
        __syncthreads();   // single-wave workgroup: S_NOP + dscnt wait (ordering only)
    }

    // ---------------- FC epilogue (from LDS) ----------------
    // fc1 + relu: this lane -> row = ln, outputs j = hi*16 + jj
    float zacc[16];
#pragma unroll
    for (int jj = 0; jj < 16; ++jj) zacc[jj] = fc1_b[hi * 16 + jj];
#pragma unroll 1
    for (int k = 0; k < kH; ++k) {
        float hv = (float)hbuf[ln * kHSH + k];
#pragma unroll
        for (int jj = 0; jj < 16; ++jj)
            zacc[jj] += hv * fc1_w[(hi * 16 + jj) * kH + k];
    }
#pragma unroll
    for (int jj = 0; jj < 16; ++jj) {
        float z = zacc[jj] > 0.0f ? zacc[jj] : 0.0f;
        zbuf[ln * 32 + hi * 16 + jj] = z;
    }
    __syncthreads();

    // fc2: lane -> (row = ln, output = hi)
    float o = fc2_b[hi];
#pragma unroll
    for (int j = 0; j < 32; ++j)
        o += zbuf[ln * 32 + j] * fc2_w[hi * 32 + j];
    out[(rowbase + ln) * 2 + hi] = o;
}

extern "C" void kernel_launch(void* const* d_in, const int* in_sizes, int n_in,
                              void* d_out, int out_size, void* d_ws, size_t ws_size,
                              hipStream_t stream) {
    const float* x     = (const float*)d_in[0];
    const float* W_ih  = (const float*)d_in[1];
    const float* W_hh  = (const float*)d_in[2];
    const float* b_ih  = (const float*)d_in[3];
    const float* b_hh  = (const float*)d_in[4];
    const float* fc1_w = (const float*)d_in[5];
    const float* fc1_b = (const float*)d_in[6];
    const float* fc2_w = (const float*)d_in[7];
    const float* fc2_b = (const float*)d_in[8];
    float* out = (float*)d_out;

    dim3 grid(kB / kRows);   // 512 single-wave workgroups
    dim3 block(32);
    lstm_wmma_kernel<<<grid, block, 0, stream>>>(x, W_ih, W_hh, b_ih, b_hh,
                                                 fc1_w, fc1_b, fc2_w, fc2_b, out);
}
